// StaticTransformerXLCore_47519518163493
// MI455X (gfx1250) — matrix operand order repaired and checked by reading, not compiled, not run on hardware
//
#include <hip/hip_runtime.h>
#include <math.h>

typedef __attribute__((ext_vector_type(16))) _Float16 v16h;
typedef __attribute__((ext_vector_type(8)))  _Float16 v8h;
typedef __attribute__((ext_vector_type(8)))  float    v8f;

#define D_MODEL   256
#define N_LAYER   6
#define N_HEAD    8
#define D_HEAD    64
#define D_INNER   1024
#define MEM_LEN   1600
#define CLAMP_LEN 1000
#define B_SZ      2
#define S_LEN     1024
#define INNER     512                 // N_HEAD * D_HEAD
#define KLEN      (S_LEN + MEM_LEN)   // 2624
#define WIN       1600                // effective attention window per query
#define ROWS      (S_LEN * B_SZ)      // 2048 activation rows, layout (s,b)
#define ZB        (B_SZ * N_HEAD)     // 16 batched (b,h) attention slices

// ---------------------------------------------------------------------------
// C[M,N] = A[M,K] * B[N,K]^T, all operands f16 (pre-converted), f32 accumulate.
// Each wave computes a 16x64 strip: one A fragment -> 4 WMMAs (4x A reuse).
// Block (32,4) covers a 64x64 C tile. grid = (N/64, M/64, batch).
// Optional f32 and/or f16 output (same layout), bias + ReLU epilogue.
// ---------------------------------------------------------------------------
__global__ void __launch_bounds__(128)
gemm_f16(const _Float16* __restrict__ A, int lda, long long sA,
         const _Float16* __restrict__ Bm, int ldb, long long sB,
         float* __restrict__ C, _Float16* __restrict__ Ch,
         int ldc, long long sC, int K,
         const float* __restrict__ bias, int relu)
{
  const int lane = threadIdx.x;            // full wave, EXEC all ones
  const int half = lane >> 4;
  const int l15  = lane & 15;
  const int m0 = (blockIdx.y << 6) + (threadIdx.y << 4);
  const int n0 = blockIdx.x << 6;

  const _Float16* Ab = A  + (long long)blockIdx.z * sA;
  const _Float16* Bb = Bm + (long long)blockIdx.z * sB;

  // A fragment (16x32): M = m0+l15, K = half*8+{0..7} U 16+half*8+{0..7}
  const _Float16* ap = Ab + (long long)(m0 + l15) * lda + (half << 3);
  // B fragment j (32x16): N = n0+16j+l15, K = half*8+{0..7} U 16+half*8+{0..7}
  const _Float16* bp = Bb + (long long)(n0 + l15) * ldb + (half << 3);
  const long long bstep = (long long)ldb << 4;   // 16 B-rows

  v8f acc[4] = {{}, {}, {}, {}};
  for (int k = 0; k < K; k += 32) {
    if (k + 32 < K) {                    // speculative prefetch of next K slice
      __builtin_prefetch(ap + 32, 0, 1);
      __builtin_prefetch(bp + 32, 0, 1);
    }
    v8h alo = *(const v8h*)(ap);
    v8h ahi = *(const v8h*)(ap + 16);
    v16h a = __builtin_shufflevector(alo, ahi,
               0, 1, 2, 3, 4, 5, 6, 7, 8, 9, 10, 11, 12, 13, 14, 15);
    v16h b0 = __builtin_shufflevector(*(const v8h*)(bp), *(const v8h*)(bp + 16),
               0, 1, 2, 3, 4, 5, 6, 7, 8, 9, 10, 11, 12, 13, 14, 15);
    v16h b1 = __builtin_shufflevector(*(const v8h*)(bp + bstep), *(const v8h*)(bp + bstep + 16),
               0, 1, 2, 3, 4, 5, 6, 7, 8, 9, 10, 11, 12, 13, 14, 15);
    v16h b2 = __builtin_shufflevector(*(const v8h*)(bp + 2 * bstep), *(const v8h*)(bp + 2 * bstep + 16),
               0, 1, 2, 3, 4, 5, 6, 7, 8, 9, 10, 11, 12, 13, 14, 15);
    v16h b3 = __builtin_shufflevector(*(const v8h*)(bp + 3 * bstep), *(const v8h*)(bp + 3 * bstep + 16),
               0, 1, 2, 3, 4, 5, 6, 7, 8, 9, 10, 11, 12, 13, 14, 15);
    acc[0] = __builtin_amdgcn_wmma_f32_16x16x32_f16(false, a, false, b0, (short)0, acc[0], false, false);
    acc[1] = __builtin_amdgcn_wmma_f32_16x16x32_f16(false, a, false, b1, (short)0, acc[1], false, false);
    acc[2] = __builtin_amdgcn_wmma_f32_16x16x32_f16(false, a, false, b2, (short)0, acc[2], false, false);
    acc[3] = __builtin_amdgcn_wmma_f32_16x16x32_f16(false, a, false, b3, (short)0, acc[3], false, false);
    ap += 32;
    bp += 32;
  }

  // D layout per tile: lane = column N, rows M = m0 + half*8 + r
  const int mrow = m0 + (half << 3);
  const long long zoff = (long long)blockIdx.z * sC;
#pragma unroll
  for (int j = 0; j < 4; ++j) {
    const int n = n0 + (j << 4) + l15;
    const float bv = bias ? bias[n] : 0.0f;
    const long long base = zoff + (long long)mrow * ldc + n;
#pragma unroll
    for (int r = 0; r < 8; ++r) {
      float v = acc[j][r] + bv;
      if (relu) v = fmaxf(v, 0.0f);
      if (C)  C[base + (long long)r * ldc]  = v;
      if (Ch) Ch[base + (long long)r * ldc] = (_Float16)v;
    }
  }
}

// ---------------------------------------------------------------------------
// Elementwise / layout kernels (f32 math, f16 shadows for GEMM operands)
// ---------------------------------------------------------------------------
__global__ void cvt_f16(const float* __restrict__ s, _Float16* __restrict__ d,
                        long long n)
{
  long long i = (long long)blockIdx.x * 256 + threadIdx.x;
  if (i < n) d[i] = (_Float16)s[i];
}

__global__ void transpose_in(const float* __restrict__ x, float* __restrict__ w,
                             _Float16* __restrict__ wh)
{ // x (B,S,D) -> w ((s,b),D) + f16 shadow
  int s = blockIdx.x, b = blockIdx.y, d = threadIdx.x;
  float v = x[((long long)b * S_LEN + s) * D_MODEL + d];
  long long o = (long long)(s * B_SZ + b) * D_MODEL + d;
  w[o]  = v;
  wh[o] = (_Float16)v;
}

__global__ void transpose_out(const float* __restrict__ w, float* __restrict__ o)
{
  int s = blockIdx.x, b = blockIdx.y, d = threadIdx.x;
  o[((long long)b * S_LEN + s) * D_MODEL + d] =
      w[(long long)(s * B_SZ + b) * D_MODEL + d];
}

__global__ void pos_emb_kernel(_Float16* __restrict__ pe)
{ // pe (KLEN, D) f16: [sin(pos*f) | cos(pos*f)], pos clamped to 1000
  int j = blockIdx.x, d = threadIdx.x;
  float pos  = fminf((float)(KLEN - 1 - j), (float)CLAMP_LEN);
  int   c    = d & 127;
  float freq = powf(10000.0f, -((float)(2 * c)) / (float)D_MODEL);
  float a    = pos * freq;
  pe[(long long)j * D_MODEL + d] = (_Float16)((d < 128) ? sinf(a) : cosf(a));
}

__global__ void split_heads(const float* __restrict__ heads,
                            const float* __restrict__ rwb,
                            const float* __restrict__ rrb,
                            _Float16* __restrict__ qrw, _Float16* __restrict__ qrr,
                            _Float16* __restrict__ kk,  _Float16* __restrict__ vT)
{ // heads (ROWS, 3*INNER) f32 -> per (b,h): q+biases, k (S,64), v^T (64,S), f16
  int i = blockIdx.x, z = blockIdx.y, d = threadIdx.x;
  int b = z >> 3, h = z & 7;
  long long row = (long long)(i * B_SZ + b) * (3 * INNER);
  int col = h * D_HEAD + d;
  float q  = heads[row + col];
  float kv = heads[row + INNER + col];
  float vv = heads[row + 2 * INNER + col];
  long long o = ((long long)z * S_LEN + i) * D_HEAD + d;
  qrw[o] = (_Float16)(q + rwb[h * D_HEAD + d]);
  qrr[o] = (_Float16)(q + rrb[h * D_HEAD + d]);
  kk[o]  = (_Float16)kv;
  vT[((long long)z * D_HEAD + d) * S_LEN + i] = (_Float16)vv;
}

__global__ void repack_rk(const _Float16* __restrict__ rk,
                          _Float16* __restrict__ rkh)
{ // rk rows 1024..2623 -> rkh (ZB, WIN, 64), replicated over batch
  int p = blockIdx.x, h = blockIdx.y, d = threadIdx.x;
  _Float16 v = rk[(long long)(1024 + p) * INNER + h * D_HEAD + d];
#pragma unroll
  for (int b = 0; b < B_SZ; ++b)
    rkh[(((long long)(b * N_HEAD + h)) * WIN + p) * D_HEAD + d] = v;
}

__global__ void merge_heads(const float* __restrict__ oh,
                            _Float16* __restrict__ vec)
{ // oh (ZB,S,64) f32 -> vec ((s,b), INNER) f16
  int i = blockIdx.x, b = blockIdx.y, c = threadIdx.x;
  int h = c >> 6, d = c & 63;
  vec[(long long)(i * B_SZ + b) * INNER + c] =
      (_Float16)oh[((long long)(b * N_HEAD + h) * S_LEN + i) * D_HEAD + d];
}

// Windowed softmax over the 1600-key window j in [i+1, i+1600].
// t = j-i-1; real keys (t >= 1599-i) get AC + BD, phantom mem keys get BD only
// (zero values => denominator-only). Writes causal prob matrix P in f16.
__global__ void attn_softmax(const float* __restrict__ AC,
                             const float* __restrict__ BD,
                             _Float16* __restrict__ P)
{
  __shared__ float red[256];
  const int i = blockIdx.x, z = blockIdx.y, tid = threadIdx.x;
  const float* bdrow = BD + ((long long)z * S_LEN + i) * WIN;
  const float* acrow = AC + ((long long)z * S_LEN + i) * S_LEN;
  _Float16*    prow  = P  + ((long long)z * S_LEN + i) * S_LEN;
  const int tr = WIN - 1 - i;          // t >= tr => real key jr = t - tr
  const float scale = 0.125f;          // 1/sqrt(64)

  float mx = -INFINITY;
  for (int t = tid; t < WIN; t += 256) {
    float s = bdrow[t];
    if (t >= tr) s += acrow[t - tr];
    mx = fmaxf(mx, s * scale);
  }
  red[tid] = mx; __syncthreads();
  for (int o = 128; o > 0; o >>= 1) {
    if (tid < o) red[tid] = fmaxf(red[tid], red[tid + o]);
    __syncthreads();
  }
  mx = red[0]; __syncthreads();

  float sum = 0.0f;
  for (int t = tid; t < WIN; t += 256) {
    float s = bdrow[t];
    if (t >= tr) s += acrow[t - tr];
    sum += expf(s * scale - mx);
  }
  red[tid] = sum; __syncthreads();
  for (int o = 128; o > 0; o >>= 1) {
    if (tid < o) red[tid] += red[tid + o];
    __syncthreads();
  }
  const float inv = 1.0f / red[0];

  for (int jr = tid; jr < S_LEN; jr += 256) {
    float pv = 0.0f;
    if (jr <= i) {
      float s = (bdrow[jr + tr] + acrow[jr]) * scale;
      pv = expf(s - mx) * inv;
    }
    prow[jr] = (_Float16)pv;
  }
}

__global__ void ln_residual(float* __restrict__ w, const float* __restrict__ delta,
                            const float* __restrict__ g, const float* __restrict__ bta,
                            _Float16* __restrict__ wh)
{
  __shared__ float red[256];
  int row = blockIdx.x, d = threadIdx.x;
  long long o = (long long)row * D_MODEL + d;
  float v = w[o] + delta[o];
  red[d] = v; __syncthreads();
  for (int s = 128; s > 0; s >>= 1) { if (d < s) red[d] += red[d + s]; __syncthreads(); }
  float mu = red[0] * (1.0f / D_MODEL); __syncthreads();
  float c = v - mu;
  red[d] = c * c; __syncthreads();
  for (int s = 128; s > 0; s >>= 1) { if (d < s) red[d] += red[d + s]; __syncthreads(); }
  float var = red[0] * (1.0f / D_MODEL);
  float r = c * rsqrtf(var + 1e-5f) * g[d] + bta[d];
  w[o]  = r;
  wh[o] = (_Float16)r;   // f16 shadow feeds the next GEMM's A operand
}

// ---------------------------------------------------------------------------
extern "C" void kernel_launch(void* const* d_in, const int* in_sizes, int n_in,
                              void* d_out, int out_size, void* d_ws, size_t ws_size,
                              hipStream_t stream)
{
  (void)in_sizes; (void)n_in; (void)out_size; (void)ws_size;
  const float* x      = (const float*)d_in[0];
  const float* qkv_w  = (const float*)d_in[1];
  const float* o_w    = (const float*)d_in[2];
  const float* r_w    = (const float*)d_in[3];
  const float* rw_b   = (const float*)d_in[4];
  const float* rr_b   = (const float*)d_in[5];
  const float* ln1_g  = (const float*)d_in[6];
  const float* ln1_b  = (const float*)d_in[7];
  const float* ff1_w  = (const float*)d_in[8];
  const float* ff1_b  = (const float*)d_in[9];
  const float* ff2_w  = (const float*)d_in[10];
  const float* ff2_b  = (const float*)d_in[11];
  const float* ln2_g  = (const float*)d_in[12];
  const float* ln2_b  = (const float*)d_in[13];
  float* out = (float*)d_out;

  char* base = (char*)d_ws;
  size_t off = 0;
  auto allocf = [&](size_t n) {
    float* r = (float*)(base + off); off += ((n * 4 + 255) & ~(size_t)255); return r;
  };
  auto alloch = [&](size_t n) {
    _Float16* r = (_Float16*)(base + off); off += ((n * 2 + 255) & ~(size_t)255); return r;
  };

  // f32 state / score buffers
  float* w     = allocf((size_t)ROWS * D_MODEL);
  float* heads = allocf((size_t)ROWS * 3 * INNER);
  float* AC    = allocf((size_t)ZB * S_LEN * S_LEN);
  float* BD    = allocf((size_t)ZB * S_LEN * WIN);
  float* oh    = allocf((size_t)ZB * S_LEN * D_HEAD);
  float* attn  = allocf((size_t)ROWS * D_MODEL);
  float* h2    = allocf((size_t)ROWS * D_MODEL);
  // f16 GEMM operands
  _Float16* wh   = alloch((size_t)ROWS * D_MODEL);
  _Float16* peh  = alloch((size_t)KLEN * D_MODEL);
  _Float16* rkf  = alloch((size_t)KLEN * INNER);
  _Float16* rkh  = alloch((size_t)ZB * WIN * D_HEAD);
  _Float16* qrw  = alloch((size_t)ZB * S_LEN * D_HEAD);
  _Float16* qrr  = alloch((size_t)ZB * S_LEN * D_HEAD);
  _Float16* kk   = alloch((size_t)ZB * S_LEN * D_HEAD);
  _Float16* vT   = alloch((size_t)ZB * D_HEAD * S_LEN);
  _Float16* P    = alloch((size_t)ZB * S_LEN * S_LEN);
  _Float16* vec  = alloch((size_t)ROWS * INNER);
  _Float16* h1   = alloch((size_t)ROWS * D_INNER);
  // f16 weights (converted once per launch)
  _Float16* qkvh = alloch((size_t)N_LAYER * 3 * INNER * D_MODEL);
  _Float16* owh  = alloch((size_t)N_LAYER * D_MODEL * INNER);
  _Float16* rwh  = alloch((size_t)N_LAYER * INNER * D_MODEL);
  _Float16* f1h  = alloch((size_t)N_LAYER * D_INNER * D_MODEL);
  _Float16* f2h  = alloch((size_t)N_LAYER * D_MODEL * D_INNER);

  auto cvt = [&](const float* s, _Float16* d, long long n) {
    cvt_f16<<<dim3((unsigned)((n + 255) / 256)), 256, 0, stream>>>(s, d, n);
  };
  cvt(qkv_w, qkvh, (long long)N_LAYER * 3 * INNER * D_MODEL);
  cvt(o_w,   owh,  (long long)N_LAYER * D_MODEL * INNER);
  cvt(r_w,   rwh,  (long long)N_LAYER * INNER * D_MODEL);
  cvt(ff1_w, f1h,  (long long)N_LAYER * D_INNER * D_MODEL);
  cvt(ff2_w, f2h,  (long long)N_LAYER * D_MODEL * D_INNER);

  const dim3 gblk(32, 4);

  transpose_in<<<dim3(S_LEN, B_SZ), D_MODEL, 0, stream>>>(x, w, wh);
  pos_emb_kernel<<<dim3(KLEN), D_MODEL, 0, stream>>>(peh);

  for (int l = 0; l < N_LAYER; ++l) {
    const _Float16* qkvl = qkvh + (size_t)l * 3 * INNER * D_MODEL;
    const _Float16* owl  = owh  + (size_t)l * D_MODEL * INNER;
    const _Float16* rwl  = rwh  + (size_t)l * INNER * D_MODEL;
    const _Float16* f1wl = f1h  + (size_t)l * D_INNER * D_MODEL;
    const _Float16* f2wl = f2h  + (size_t)l * D_MODEL * D_INNER;
    const float* rwbl = rw_b  + (size_t)l * INNER;
    const float* rrbl = rr_b  + (size_t)l * INNER;
    const float* f1bl = ff1_b + (size_t)l * D_INNER;
    const float* f2bl = ff2_b + (size_t)l * D_MODEL;

    // rk = pos_emb @ r_w^T   (KLEN x INNER), f16 out
    gemm_f16<<<dim3(INNER / 64, KLEN / 64, 1), gblk, 0, stream>>>(
        peh, D_MODEL, 0, rwl, D_MODEL, 0,
        nullptr, rkf, INNER, 0, D_MODEL, nullptr, 0);
    repack_rk<<<dim3(WIN, N_HEAD), D_HEAD, 0, stream>>>(rkf, rkh);

    // heads = w @ qkv_w^T   (ROWS x 3*INNER), f32 out (bias-add next)
    gemm_f16<<<dim3(3 * INNER / 64, ROWS / 64, 1), gblk, 0, stream>>>(
        wh, D_MODEL, 0, qkvl, D_MODEL, 0,
        heads, nullptr, 3 * INNER, 0, D_MODEL, nullptr, 0);
    split_heads<<<dim3(S_LEN, ZB), D_HEAD, 0, stream>>>(heads, rwbl, rrbl,
                                                        qrw, qrr, kk, vT);

    // AC[z] = (q + r_w_bias) @ k^T   (S x S, batched over 16 (b,h))
    gemm_f16<<<dim3(S_LEN / 64, S_LEN / 64, ZB), gblk, 0, stream>>>(
        qrw, D_HEAD, (long long)S_LEN * D_HEAD,
        kk,  D_HEAD, (long long)S_LEN * D_HEAD,
        AC, nullptr, S_LEN, (long long)S_LEN * S_LEN, D_HEAD, nullptr, 0);
    // BD[z] = (q + r_r_bias) @ rk[1024:]^T   (S x WIN)
    gemm_f16<<<dim3(WIN / 64, S_LEN / 64, ZB), gblk, 0, stream>>>(
        qrr, D_HEAD, (long long)S_LEN * D_HEAD,
        rkh, D_HEAD, (long long)WIN * D_HEAD,
        BD, nullptr, WIN, (long long)S_LEN * WIN, D_HEAD, nullptr, 0);

    attn_softmax<<<dim3(S_LEN, ZB), 256, 0, stream>>>(AC, BD, P);

    // oh[z] = P @ V   (S x 64)
    gemm_f16<<<dim3(D_HEAD / 64, S_LEN / 64, ZB), gblk, 0, stream>>>(
        P,  S_LEN, (long long)S_LEN * S_LEN,
        vT, S_LEN, (long long)D_HEAD * S_LEN,
        oh, nullptr, D_HEAD, (long long)S_LEN * D_HEAD, S_LEN, nullptr, 0);

    merge_heads<<<dim3(S_LEN, B_SZ), INNER, 0, stream>>>(oh, vec);

    // attn = vec @ o_w^T
    gemm_f16<<<dim3(D_MODEL / 64, ROWS / 64, 1), gblk, 0, stream>>>(
        vec, INNER, 0, owl, INNER, 0,
        attn, nullptr, D_MODEL, 0, INNER, nullptr, 0);
    ln_residual<<<dim3(ROWS), D_MODEL, 0, stream>>>(w, attn,
        ln1_g + (size_t)l * D_MODEL, ln1_b + (size_t)l * D_MODEL, wh);

    // FFN: h1 = relu(w @ ff1^T + b1) -> f16 directly; h2 = h1 @ ff2^T + b2
    gemm_f16<<<dim3(D_INNER / 64, ROWS / 64, 1), gblk, 0, stream>>>(
        wh, D_MODEL, 0, f1wl, D_MODEL, 0,
        nullptr, h1, D_INNER, 0, D_MODEL, f1bl, 1);
    gemm_f16<<<dim3(D_MODEL / 64, ROWS / 64, 1), gblk, 0, stream>>>(
        h1, D_INNER, 0, f2wl, D_INNER, 0,
        h2, nullptr, D_MODEL, 0, D_INNER, f2bl, 0);
    ln_residual<<<dim3(ROWS), D_MODEL, 0, stream>>>(w, h2,
        ln2_g + (size_t)l * D_MODEL, ln2_b + (size_t)l * D_MODEL, wh);
  }

  transpose_out<<<dim3(S_LEN, B_SZ), D_MODEL, 0, stream>>>(w, out);
}
